// LRU_27178553049306
// MI455X (gfx1250) — matrix operand order repaired and checked by reading, not compiled, not run on hardware
//
#include <hip/hip_runtime.h>
#include <hip/hip_bf16.h>

// Problem sizes (fixed by the reference)
#define TT   4096   // SEQ_LEN
#define DM   4096   // D_MODEL
#define HH   2048   // D_HIDDEN
#define CH   64     // scan chunk length
#define NCH  (TT / CH)

typedef _Float16 half_t;
typedef __attribute__((ext_vector_type(16))) _Float16 v16h;
typedef __attribute__((ext_vector_type(8)))  _Float16 h8;
typedef __attribute__((ext_vector_type(8)))  float    v8f;

union F16Frag { v16h v; h8 h[2]; };

#define LSTR 40   // LDS row stride in halves (32 + 8 pad -> conflict-free b128 reads)

__device__ __forceinline__ v8f wmma_f16(v16h a, v16h b, v8f c) {
  // D = A(16x32 f16) * B(32x16 f16) + C(16x16 f32)
  return __builtin_amdgcn_wmma_f32_16x16x32_f16(
      /*neg_a=*/false, a, /*neg_b=*/false, b,
      /*c_mod=*/(short)0, c, /*reuse_a=*/false, /*reuse_b=*/false);
}

// Low 32 bits of a generic pointer to LDS == wave-relative LDS byte address.
__device__ __forceinline__ unsigned lds_addr32(const void* p) {
  return (unsigned)(unsigned long long)p;
}

// Async VGPR-bypassing 16B copy global -> LDS (tracked with ASYNCcnt).
__device__ __forceinline__ void async_ld16(unsigned lds, const void* g) {
  asm volatile("global_load_async_to_lds_b128 %0, %1, off"
               :: "v"(lds), "v"((unsigned long long)g)
               : "memory");
}

__device__ __forceinline__ void wait_async0() {
  asm volatile("s_wait_asynccnt 0x0" ::: "memory");
}

// A-fragment: row m = rowBase + lane%16; lanes 0-15 hold K 0-7 & 16-23,
// lanes 16-31 hold K 8-15 & 24-31 (CDNA5 16-bit A 16x32 layout).
__device__ __forceinline__ v16h lds_a_frag(const half_t* s, int rowBase, int lane) {
  const int r  = rowBase + (lane & 15);
  const int kb = (lane >> 4) << 3;
  F16Frag f;
  f.h[0] = *(const h8*)(s + r * LSTR + kb);
  f.h[1] = *(const h8*)(s + r * LSTR + kb + 16);
  return f.v;
}

// B-fragment: col n = colBase + lane%16; lanes 0-15 hold K 0-7 & 16-23,
// lanes 16-31 hold K 8-15 & 24-31 (same as A). LDS holds B as [n][k].
__device__ __forceinline__ v16h lds_b_frag(const half_t* s, int colBase, int lane) {
  const int n  = colBase + (lane & 15);
  const int k0 = (lane >> 4) << 3;
  F16Frag f;
  f.h[0] = *(const h8*)(s + n * LSTR + k0);
  f.h[1] = *(const h8*)(s + n * LSTR + k0 + 16);
  return f.v;
}

// ---------------------------------------------------------------------------
// Per-channel constants: lambda = exp(-exp(nu) + i*exp(theta)), gamma = exp(gl)
// ---------------------------------------------------------------------------
__global__ void k_prep(const float* __restrict__ nu, const float* __restrict__ th,
                       const float* __restrict__ gl,
                       float* __restrict__ lamR, float* __restrict__ lamI,
                       float* __restrict__ gam) {
  int h = blockIdx.x * 256 + threadIdx.x;
  float r = __expf(-__expf(nu[h]));
  float t = __expf(th[h]);
  lamR[h] = r * __cosf(t);
  lamI[h] = r * __sinf(t);
  gam[h]  = __expf(gl[h]);
}

// ---------------------------------------------------------------------------
// f32 -> f16 conversions for the GEMM operands
// ---------------------------------------------------------------------------
__global__ void k_cvt_x(const float* __restrict__ x, half_t* __restrict__ xh) {
  int i = blockIdx.x * 256 + threadIdx.x;
  xh[i] = (half_t)x[i];
}

__global__ void k_cvt_b(const float* __restrict__ Br, const float* __restrict__ Bi,
                        const float* __restrict__ gam,
                        half_t* __restrict__ o0, half_t* __restrict__ o1) {
  int i = blockIdx.x * 256 + threadIdx.x;
  int h = i >> 12;              // DM == 4096
  float g = gam[h];
  o0[i] = (half_t)(Br[i] * g);
  o1[i] = (half_t)(Bi[i] * g);
}

__global__ void k_cvt_c(const float* __restrict__ Cr, const float* __restrict__ Ci,
                        half_t* __restrict__ o0, half_t* __restrict__ o1) {
  int i = blockIdx.x * 256 + threadIdx.x;
  o0[i] = (half_t)Cr[i];
  o1[i] = (half_t)Ci[i];
}

// ---------------------------------------------------------------------------
// GEMM1: Bu(T,H) = x(T,D) * Bn(H,D)^T for both re/im operands (shared A)
// Block tile 128x128, 8 waves, each wave 64x32, dual output.
// Double-buffered LDS, async global->LDS copies overlapped with WMMA.
// ---------------------------------------------------------------------------
__global__ __launch_bounds__(256)
void k_gemm_bu(const half_t* __restrict__ A,
               const half_t* __restrict__ B0, const half_t* __restrict__ B1,
               float* __restrict__ O0, float* __restrict__ O1) {
  const int N = HH, K = DM;
  __shared__ __align__(16) half_t As [2][128 * LSTR];
  __shared__ __align__(16) half_t Bs0[2][128 * LSTR];
  __shared__ __align__(16) half_t Bs1[2][128 * LSTR];
  const int tid = threadIdx.x, lane = tid & 31, wid = tid >> 5;
  const int m0 = blockIdx.y * 128;
  const int n0 = blockIdx.x * 128;
  const int wm = (wid >> 2) * 64;     // 2 wave-rows
  const int wn = (wid & 3) * 32;      // 4 wave-cols

  // Issue one K-tile's async copies (6 x b128 per thread, ASYNCcnt <= 6).
  auto issue = [&](int buf, int kk) {
    const unsigned aoff  = lds_addr32(&As [buf][0]);
    const unsigned b0off = lds_addr32(&Bs0[buf][0]);
    const unsigned b1off = lds_addr32(&Bs1[buf][0]);
    for (int i = tid; i < 128 * 4; i += 256) {
      int r = i >> 2, sg = (i & 3) << 3;
      unsigned l = (unsigned)(r * LSTR + sg) * (unsigned)sizeof(half_t);
      async_ld16(aoff  + l, A  + (size_t)(m0 + r) * K + kk + sg);
      async_ld16(b0off + l, B0 + (size_t)(n0 + r) * K + kk + sg);
      async_ld16(b1off + l, B1 + (size_t)(n0 + r) * K + kk + sg);
    }
  };

  const v8f vz = {0.f,0.f,0.f,0.f,0.f,0.f,0.f,0.f};
  v8f acc[2][4][2];
  for (int o = 0; o < 2; ++o)
    for (int a = 0; a < 4; ++a)
      for (int b = 0; b < 2; ++b) acc[o][a][b] = vz;

  issue(0, 0);
  for (int kk = 0; kk < K; kk += 32) {
    const int cur = (kk >> 5) & 1;
    wait_async0();        // my copies for buf[cur] have landed in LDS
    __syncthreads();      // everyone's copies landed; prev compute done
    if (kk + 32 < K)      // overlap next tile's copies with this tile's WMMAs
      issue(cur ^ 1, kk + 32);

    v16h af[4], b0f[2], b1f[2];
    for (int t = 0; t < 4; ++t) af[t] = lds_a_frag(As[cur], wm + t * 16, lane);
    for (int t = 0; t < 2; ++t) {
      b0f[t] = lds_b_frag(Bs0[cur], wn + t * 16, lane);
      b1f[t] = lds_b_frag(Bs1[cur], wn + t * 16, lane);
    }
    for (int tm = 0; tm < 4; ++tm)
      for (int tn = 0; tn < 2; ++tn) {
        acc[0][tm][tn] = wmma_f16(af[tm], b0f[tn], acc[0][tm][tn]);
        acc[1][tm][tn] = wmma_f16(af[tm], b1f[tn], acc[1][tm][tn]);
      }
  }

  // C/D layout: VGPR j holds M = j (+8 for lanes 16-31), N = lane%16
  for (int tm = 0; tm < 4; ++tm)
    for (int tn = 0; tn < 2; ++tn) {
      int rb  = m0 + wm + tm * 16 + ((lane >> 4) << 3);
      int col = n0 + wn + tn * 16 + (lane & 15);
      for (int j = 0; j < 8; ++j) {
        O0[(size_t)(rb + j) * N + col] = acc[0][tm][tn][j];
        O1[(size_t)(rb + j) * N + col] = acc[1][tm][tn][j];
      }
    }
}

// ---------------------------------------------------------------------------
// Chunked associative scan:  h_t = lambda*h_{t-1} + Bu_t   (complex, per channel)
// Pass A: local inclusive scans per 64-step chunk (in place), emit chunk carry.
// Pass B: per-channel exclusive prefix over 64 chunk carries with lambda^64.
// Pass C: h[t] = local[t] + lambda^{k+1} * P[chunk]; convert to f16 for GEMM2.
// ---------------------------------------------------------------------------
__global__ void k_scan_a(float* __restrict__ BuR, float* __restrict__ BuI,
                         const float* __restrict__ lamR, const float* __restrict__ lamI,
                         float* __restrict__ carR, float* __restrict__ carI) {
  int h = blockIdx.y * 256 + threadIdx.x;
  int c = blockIdx.x;
  float lr = lamR[h], li = lamI[h];
  float ar = 0.f, ai = 0.f;
  int t0 = c * CH;
  for (int k = 0; k < CH; ++k) {
    size_t idx = (size_t)(t0 + k) * HH + h;
    float br = BuR[idx], bi = BuI[idx];
    float nr = lr * ar - li * ai + br;
    float ni = lr * ai + li * ar + bi;
    ar = nr; ai = ni;
    BuR[idx] = ar; BuI[idx] = ai;      // overwrite with local scan
  }
  carR[(size_t)c * HH + h] = ar;
  carI[(size_t)c * HH + h] = ai;
}

__global__ void k_scan_b(float* __restrict__ carR, float* __restrict__ carI,
                         const float* __restrict__ lamR, const float* __restrict__ lamI) {
  int h = blockIdx.x * 256 + threadIdx.x;
  float lr = lamR[h], li = lamI[h];
  // lambda^CH via 6 complex squarings (CH = 64 = 2^6)
  float pr = lr, pi = li;
  for (int s = 0; s < 6; ++s) {
    float nr = pr * pr - pi * pi, ni = 2.f * pr * pi;
    pr = nr; pi = ni;
  }
  float er = 0.f, ei = 0.f;            // exclusive prefix (state before chunk c)
  for (int c = 0; c < NCH; ++c) {
    size_t idx = (size_t)c * HH + h;
    float cr = carR[idx], ci = carI[idx];
    carR[idx] = er; carI[idx] = ei;
    float nr = pr * er - pi * ei + cr;
    float ni = pr * ei + pi * er + ci;
    er = nr; ei = ni;
  }
}

__global__ void k_scan_c(const float* __restrict__ BuR, const float* __restrict__ BuI,
                         const float* __restrict__ carR, const float* __restrict__ carI,
                         const float* __restrict__ lamR, const float* __restrict__ lamI,
                         half_t* __restrict__ hR, half_t* __restrict__ hI) {
  int h = blockIdx.y * 256 + threadIdx.x;
  int c = blockIdx.x;
  float lr = lamR[h], li = lamI[h];
  float er = carR[(size_t)c * HH + h];
  float ei = carI[(size_t)c * HH + h];
  float fr = lr, fi = li;              // lambda^{k+1}
  int t0 = c * CH;
  for (int k = 0; k < CH; ++k) {
    size_t idx = (size_t)(t0 + k) * HH + h;
    float vr = BuR[idx] + fr * er - fi * ei;
    float vi = BuI[idx] + fr * ei + fi * er;
    hR[idx] = (half_t)vr;
    hI[idx] = (half_t)vi;
    float nr = fr * lr - fi * li;
    float ni = fr * li + fi * lr;
    fr = nr; fi = ni;
  }
}

// ---------------------------------------------------------------------------
// GEMM2: out(T,D) = hidRe*CRe^T - hidIm*CIm^T + x .* Dvec
// Block tile 128x64, 8 waves, each wave 64x16, dual A (re/im) operands.
// Double-buffered LDS with async copies, skip-connection epilogue.
// ---------------------------------------------------------------------------
__global__ __launch_bounds__(256)
void k_gemm_out(const half_t* __restrict__ A0, const half_t* __restrict__ A1,
                const half_t* __restrict__ B0, const half_t* __restrict__ B1,
                const float* __restrict__ X, const float* __restrict__ Dv,
                float* __restrict__ Out) {
  const int N = DM, K = HH;
  __shared__ __align__(16) half_t As0[2][128 * LSTR];
  __shared__ __align__(16) half_t As1[2][128 * LSTR];
  __shared__ __align__(16) half_t Bs0[2][ 64 * LSTR];
  __shared__ __align__(16) half_t Bs1[2][ 64 * LSTR];
  const int tid = threadIdx.x, lane = tid & 31, wid = tid >> 5;
  const int m0 = blockIdx.y * 128;
  const int n0 = blockIdx.x * 64;
  const int wm = (wid >> 2) * 64;
  const int wn = (wid & 3) * 16;

  auto issue = [&](int buf, int kk) {
    const unsigned a0off = lds_addr32(&As0[buf][0]);
    const unsigned a1off = lds_addr32(&As1[buf][0]);
    const unsigned b0off = lds_addr32(&Bs0[buf][0]);
    const unsigned b1off = lds_addr32(&Bs1[buf][0]);
    for (int i = tid; i < 128 * 4; i += 256) {
      int r = i >> 2, sg = (i & 3) << 3;
      unsigned l = (unsigned)(r * LSTR + sg) * (unsigned)sizeof(half_t);
      async_ld16(a0off + l, A0 + (size_t)(m0 + r) * K + kk + sg);
      async_ld16(a1off + l, A1 + (size_t)(m0 + r) * K + kk + sg);
    }
    for (int i = tid; i < 64 * 4; i += 256) {
      int r = i >> 2, sg = (i & 3) << 3;
      unsigned l = (unsigned)(r * LSTR + sg) * (unsigned)sizeof(half_t);
      async_ld16(b0off + l, B0 + (size_t)(n0 + r) * K + kk + sg);
      async_ld16(b1off + l, B1 + (size_t)(n0 + r) * K + kk + sg);
    }
  };

  const v8f vz = {0.f,0.f,0.f,0.f,0.f,0.f,0.f,0.f};
  v8f accS[4], accT[4];
  for (int t = 0; t < 4; ++t) { accS[t] = vz; accT[t] = vz; }

  issue(0, 0);
  for (int kk = 0; kk < K; kk += 32) {
    const int cur = (kk >> 5) & 1;
    wait_async0();
    __syncthreads();
    if (kk + 32 < K)
      issue(cur ^ 1, kk + 32);

    v16h b0f = lds_b_frag(Bs0[cur], wn, lane);
    v16h b1f = lds_b_frag(Bs1[cur], wn, lane);
    for (int tm = 0; tm < 4; ++tm) {
      v16h a0 = lds_a_frag(As0[cur], wm + tm * 16, lane);
      v16h a1 = lds_a_frag(As1[cur], wm + tm * 16, lane);
      accS[tm] = wmma_f16(a0, b0f, accS[tm]);
      accT[tm] = wmma_f16(a1, b1f, accT[tm]);
    }
  }

  int col = n0 + wn + (lane & 15);
  float dv = Dv[col];
  for (int tm = 0; tm < 4; ++tm) {
    int rb = m0 + wm + tm * 16 + ((lane >> 4) << 3);
    for (int j = 0; j < 8; ++j) {
      size_t idx = (size_t)(rb + j) * N + col;
      Out[idx] = accS[tm][j] - accT[tm][j] + X[idx] * dv;
    }
  }
}

// ---------------------------------------------------------------------------
extern "C" void kernel_launch(void* const* d_in, const int* in_sizes, int n_in,
                              void* d_out, int out_size, void* d_ws, size_t ws_size,
                              hipStream_t stream) {
  const float* x   = (const float*)d_in[0];  // (T, D)
  const float* nu  = (const float*)d_in[1];  // (H,)
  const float* th  = (const float*)d_in[2];  // (H,)
  const float* gl  = (const float*)d_in[3];  // (H,)
  const float* Bre = (const float*)d_in[4];  // (H, D)
  const float* Bim = (const float*)d_in[5];  // (H, D)
  const float* Cre = (const float*)d_in[6];  // (D, H)
  const float* Cim = (const float*)d_in[7];  // (D, H)
  const float* Dv  = (const float*)d_in[8];  // (D,)
  float* out = (float*)d_out;                // (T, D) f32

  char* w = (char*)d_ws;
  size_t o = 0;
  half_t* xh  = (half_t*)(w + o); o += (size_t)TT * DM * sizeof(half_t);
  half_t* bn0 = (half_t*)(w + o); o += (size_t)HH * DM * sizeof(half_t);
  half_t* bn1 = (half_t*)(w + o); o += (size_t)HH * DM * sizeof(half_t);
  half_t* c0  = (half_t*)(w + o); o += (size_t)DM * HH * sizeof(half_t);
  half_t* c1  = (half_t*)(w + o); o += (size_t)DM * HH * sizeof(half_t);
  float*  buR = (float*)(w + o);  o += (size_t)TT * HH * sizeof(float);
  float*  buI = (float*)(w + o);  o += (size_t)TT * HH * sizeof(float);
  half_t* hR  = (half_t*)(w + o); o += (size_t)TT * HH * sizeof(half_t);
  half_t* hI  = (half_t*)(w + o); o += (size_t)TT * HH * sizeof(half_t);
  float* lamR = (float*)(w + o);  o += (size_t)HH * sizeof(float);
  float* lamI = (float*)(w + o);  o += (size_t)HH * sizeof(float);
  float* gam  = (float*)(w + o);  o += (size_t)HH * sizeof(float);
  float* carR = (float*)(w + o);  o += (size_t)NCH * HH * sizeof(float);
  float* carI = (float*)(w + o);  o += (size_t)NCH * HH * sizeof(float);
  (void)o; (void)ws_size; (void)in_sizes; (void)n_in; (void)out_size;

  k_prep<<<HH / 256, 256, 0, stream>>>(nu, th, gl, lamR, lamI, gam);
  k_cvt_x<<<(TT * DM) / 256, 256, 0, stream>>>(x, xh);
  k_cvt_b<<<(HH * DM) / 256, 256, 0, stream>>>(Bre, Bim, gam, bn0, bn1);
  k_cvt_c<<<(DM * HH) / 256, 256, 0, stream>>>(Cre, Cim, c0, c1);

  k_gemm_bu<<<dim3(HH / 128, TT / 128), 256, 0, stream>>>(xh, bn0, bn1, buR, buI);

  k_scan_a<<<dim3(NCH, HH / 256), 256, 0, stream>>>(buR, buI, lamR, lamI, carR, carI);
  k_scan_b<<<HH / 256, 256, 0, stream>>>(carR, carI, lamR, lamI);
  k_scan_c<<<dim3(NCH, HH / 256), 256, 0, stream>>>(buR, buI, carR, carI, lamR, lamI, hR, hI);

  k_gemm_out<<<dim3(DM / 64, TT / 128), 256, 0, stream>>>(hR, hI, c0, c1, x, Dv, out);
}
